// MultiRigidSidechain_23476291240026
// MI455X (gfx1250) — matrix operand order repaired and checked by reading, not compiled, not run on hardware
//
#include <hip/hip_runtime.h>
#include <stdint.h>

// ---------------------------------------------------------------------------
// MultiRigidSidechain fused kernel for gfx1250 (MI455X).
// GEMMs run on v_wmma_f32_16x16x32_f16 (f32 accumulate); everything is fused
// into one kernel so the 128-wide activation never touches HBM.
// ---------------------------------------------------------------------------

#define N_MSA   384
#define N_HID   128

typedef __attribute__((ext_vector_type(16))) _Float16 v16h;
typedef __attribute__((ext_vector_type(8)))  float    v8f;

union Frag16 { v16h v; _Float16 h[16]; uint32_t u[8]; };

__device__ __forceinline__ v8f wmma16(v16h a, v16h b, v8f c) {
  // D = A(16x32 f16) * B(32x16 f16) + C(16x16 f32)
  return __builtin_amdgcn_wmma_f32_16x16x32_f16(
      /*neg_a=*/false, a, /*neg_b=*/false, b,
      /*c_mod=*/(short)0, c, /*reuse_a=*/false, /*reuse_b=*/false);
}

// A-matrix (16x32 f16) VGPR layout: VGPR i holds K pair starting at
// (i&4 ? 16 : 0) + 2*(i&3) + (lane>=16 ? 8 : 0), row M = lane&15.
__device__ __forceinline__ int a_koff(int i, int hs) {
  return ((i & 4) ? 16 : 0) + ((i & 3) << 1) + (hs ? 8 : 0);
}

// Build A fragment from global fp32 row with fused relu + cvt-to-f16.
__device__ __forceinline__ v16h loadA_relu_g(const float* row, int kb, int hs) {
  Frag16 f;
#pragma unroll
  for (int i = 0; i < 8; ++i) {
    int k = kb + a_koff(i, hs);
    float2 p = *reinterpret_cast<const float2*>(row + k);
    f.h[2 * i]     = (_Float16)fmaxf(p.x, 0.0f);
    f.h[2 * i + 1] = (_Float16)fmaxf(p.y, 0.0f);
  }
  return f.v;
}

// Build A fragment from a per-wave LDS tile [16][128] f16 (already relu'd).
__device__ __forceinline__ v16h loadA_lds(const _Float16* sact, int m, int kb, int hs) {
  Frag16 f;
#pragma unroll
  for (int i = 0; i < 8; ++i) {
    int k = kb + a_koff(i, hs);
    f.u[i] = *reinterpret_cast<const uint32_t*>(sact + m * N_HID + k);
  }
  return f.v;
}

// B-matrix (32x16 f16) layout: same K order as A (a_koff),
// N = lane&15; VGPR i holds the (k, k+1) pair. Weights stored transposed
// N-major [N][K] f16 so each pair is one aligned dword.
__device__ __forceinline__ v16h loadB_wt(const _Float16* wrow, int kb, int hs) {
  Frag16 f;
#pragma unroll
  for (int i = 0; i < 8; ++i) {
    int k = kb + a_koff(i, hs);
    f.u[i] = *reinterpret_cast<const uint32_t*>(wrow + k);
  }
  return f.v;
}

// C/D layout: VGPR j = element (M = j + 8*half, N = lane&15 + 16*nt).
__device__ __forceinline__ void store_relu_lds(_Float16* sact, const v8f (&acc)[8],
                                               int hs, int nloc) {
#pragma unroll
  for (int nt = 0; nt < 8; ++nt)
#pragma unroll
    for (int j = 0; j < 8; ++j)
      sact[(j + 8 * hs) * N_HID + nt * 16 + nloc] = (_Float16)fmaxf(acc[nt][j], 0.0f);
}

__device__ __forceinline__ void mat3mul(float* D, const float* A, const float* B) {
#pragma unroll
  for (int r = 0; r < 3; ++r)
#pragma unroll
    for (int c = 0; c < 3; ++c)
      D[r * 3 + c] = A[r * 3 + 0] * B[0 * 3 + c] + A[r * 3 + 1] * B[1 * 3 + c] +
                     A[r * 3 + 2] * B[2 * 3 + c];
}

// ---------------- workspace layout (f16 elements) ----------------
#define W0T_OFF   0          // [128][384]
#define W1T_OFF   49152      // [128][384]
#define WR1T_OFF  98304      // [2][128][128]
#define WR2T_OFF  131072     // [2][128][128]
#define WANGT_OFF 163840     // [16][128] (rows 14,15 zero)
#define WS_TOTAL  165888

__global__ void prep_weights_kernel(const float* __restrict__ w_in0,
                                    const float* __restrict__ w_in1,
                                    const float* __restrict__ Wr1,
                                    const float* __restrict__ Wr2,
                                    const float* __restrict__ w_ang,
                                    _Float16* __restrict__ ws) {
  int t = blockIdx.x * blockDim.x + threadIdx.x;
  if (t >= WS_TOTAL) return;
  if (t < W1T_OFF) {
    int n = t / N_MSA, k = t % N_MSA;
    ws[t] = (_Float16)w_in0[k * N_HID + n];
  } else if (t < WR1T_OFF) {
    int u = t - W1T_OFF; int n = u / N_MSA, k = u % N_MSA;
    ws[t] = (_Float16)w_in1[k * N_HID + n];
  } else if (t < WR2T_OFF) {
    int u = t - WR1T_OFF; int i = u / 16384; int rem = u % 16384;
    int n = rem / N_HID, k = rem % N_HID;
    ws[t] = (_Float16)Wr1[i * 16384 + k * N_HID + n];
  } else if (t < WANGT_OFF) {
    int u = t - WR2T_OFF; int i = u / 16384; int rem = u % 16384;
    int n = rem / N_HID, k = rem % N_HID;
    ws[t] = (_Float16)Wr2[i * 16384 + k * N_HID + n];
  } else {
    int u = t - WANGT_OFF; int n = u / N_HID, k = u % N_HID;
    ws[t] = (n < 14) ? (_Float16)w_ang[k * 14 + n] : (_Float16)0.0f;
  }
}

__global__ __launch_bounds__(256) void sidechain_fused_kernel(
    const float* __restrict__ affine, const float* __restrict__ rep0,
    const float* __restrict__ rep1, const int* __restrict__ aatype,
    const float* __restrict__ b_in0, const float* __restrict__ b_in1,
    const float* __restrict__ br1, const float* __restrict__ br2,
    const float* __restrict__ b_ang, const float* __restrict__ dframes,
    const int* __restrict__ gidx, const float* __restrict__ amask,
    const float* __restrict__ lpos, const _Float16* __restrict__ ws,
    float* __restrict__ out, int n_res) {
  __shared__ _Float16 s_act[8][16 * N_HID];  // per-wave relu(act) tile, f16
  __shared__ float    s_un[8][256];          // per-wave unnorm angles, f32

  const int wave = threadIdx.x >> 5;
  const int lane = threadIdx.x & 31;
  const int hs   = lane >> 4;     // half-select: which 16-lane group
  const int nloc = lane & 15;     // N (and A-row) index within tile
  const int rowBase = blockIdx.x * 128 + wave * 16;

  const _Float16* w0T  = ws + W0T_OFF;
  const _Float16* w1T  = ws + W1T_OFF;
  const _Float16* Wr1T = ws + WR1T_OFF;
  const _Float16* Wr2T = ws + WR2T_OFF;
  const _Float16* wAT  = ws + WANGT_OFF;

  _Float16* sact = s_act[wave];
  float*    sun  = s_un[wave];

  // output section offsets (floats): angles, unnorm, pos, rot_g, trans_g
  const long o_angles = 0;
  const long o_unnorm = (long)n_res * 14;
  const long o_pos    = o_unnorm + (long)n_res * 14;
  const long o_rot    = o_pos + (long)n_res * 42;
  const long o_trans  = o_rot + (long)n_res * 72;

  // ---------------- Phase A: act = relu(rep0)@W0 + b0 + relu(rep1)@W1 + b1
  v8f acc[8];
#pragma unroll
  for (int nt = 0; nt < 8; ++nt) {
    float b = b_in0[nt * 16 + nloc] + b_in1[nt * 16 + nloc];
#pragma unroll
    for (int j = 0; j < 8; ++j) acc[nt][j] = b;
  }
  const float* r0row = rep0 + (long)(rowBase + nloc) * N_MSA;
  const float* r1row = rep1 + (long)(rowBase + nloc) * N_MSA;
  for (int kc = 0; kc < N_MSA / 32; ++kc) {
    int kb = kc * 32;
    v16h a0 = loadA_relu_g(r0row, kb, hs);
    v16h a1 = loadA_relu_g(r1row, kb, hs);
#pragma unroll
    for (int nt = 0; nt < 8; ++nt) {
      int n = nt * 16 + nloc;
      acc[nt] = wmma16(a0, loadB_wt(w0T + n * N_MSA, kb, hs), acc[nt]);
      acc[nt] = wmma16(a1, loadB_wt(w1T + n * N_MSA, kb, hs), acc[nt]);
    }
  }

  // ---------------- Residual blocks
  for (int blk = 0; blk < 2; ++blk) {
    v8f old[8];
#pragma unroll
    for (int nt = 0; nt < 8; ++nt) old[nt] = acc[nt];

    store_relu_lds(sact, acc, hs, nloc);
    __syncthreads();
#pragma unroll
    for (int nt = 0; nt < 8; ++nt) {
      float b = br1[blk * N_HID + nt * 16 + nloc];
#pragma unroll
      for (int j = 0; j < 8; ++j) acc[nt][j] = b;
    }
    for (int kc = 0; kc < 4; ++kc) {
      int kb = kc * 32;
      v16h a = loadA_lds(sact, nloc, kb, hs);
#pragma unroll
      for (int nt = 0; nt < 8; ++nt)
        acc[nt] = wmma16(a, loadB_wt(Wr1T + blk * 16384 + (nt * 16 + nloc) * N_HID, kb, hs),
                         acc[nt]);
    }
    __syncthreads();
    store_relu_lds(sact, acc, hs, nloc);
    __syncthreads();
#pragma unroll
    for (int nt = 0; nt < 8; ++nt) {
      float b = br2[blk * N_HID + nt * 16 + nloc];
#pragma unroll
      for (int j = 0; j < 8; ++j) acc[nt][j] = b;
    }
    for (int kc = 0; kc < 4; ++kc) {
      int kb = kc * 32;
      v16h a = loadA_lds(sact, nloc, kb, hs);
#pragma unroll
      for (int nt = 0; nt < 8; ++nt)
        acc[nt] = wmma16(a, loadB_wt(Wr2T + blk * 16384 + (nt * 16 + nloc) * N_HID, kb, hs),
                         acc[nt]);
    }
#pragma unroll
    for (int nt = 0; nt < 8; ++nt) acc[nt] += old[nt];  // residual
    __syncthreads();
  }

  // ---------------- Angles GEMM: unnorm = relu(act)@w_ang + b_ang (N=14 pad 16)
  store_relu_lds(sact, acc, hs, nloc);
  __syncthreads();
  v8f accA;
  {
    float b = (nloc < 14) ? b_ang[nloc] : 0.0f;
#pragma unroll
    for (int j = 0; j < 8; ++j) accA[j] = b;
  }
  for (int kc = 0; kc < 4; ++kc) {
    int kb = kc * 32;
    v16h a = loadA_lds(sact, nloc, kb, hs);
    accA = wmma16(a, loadB_wt(wAT + nloc * N_HID, kb, hs), accA);
  }
#pragma unroll
  for (int j = 0; j < 8; ++j) sun[(j + 8 * hs) * 16 + nloc] = accA[j];
  __syncthreads();

  // ---------------- Geometry: one residue per lane (lanes 0..15)
  if (lane < 16) {
    const int r = rowBase + lane;
    float u[14];
#pragma unroll
    for (int c = 0; c < 14; ++c) u[c] = sun[lane * 16 + c];

    float sn[8], cs[8];
    sn[0] = 0.0f; cs[0] = 1.0f;
#pragma unroll
    for (int a = 0; a < 7; ++a) {
      float s = u[2 * a], c = u[2 * a + 1];
      out[o_unnorm + (long)r * 14 + 2 * a]     = s;
      out[o_unnorm + (long)r * 14 + 2 * a + 1] = c;
      float inv = 1.0f / sqrtf(fmaxf(s * s + c * c, 1e-12f));
      float sa = s * inv, ca = c * inv;
      out[o_angles + (long)r * 14 + 2 * a]     = sa;
      out[o_angles + (long)r * 14 + 2 * a + 1] = ca;
      sn[a + 1] = sa; cs[a + 1] = ca;
    }

    // backbone frame from quaternion
    const float* af = affine + (long)r * 7;
    float qw = af[0], qx = af[1], qy = af[2], qz = af[3];
    float qn = 1.0f / sqrtf(fmaxf(qw * qw + qx * qx + qy * qy + qz * qz, 1e-12f));
    qw *= qn; qx *= qn; qy *= qn; qz *= qn;
    float Rb[9];
    Rb[0] = 1.f - 2.f * (qy * qy + qz * qz); Rb[1] = 2.f * (qx * qy - qw * qz); Rb[2] = 2.f * (qx * qz + qw * qy);
    Rb[3] = 2.f * (qx * qy + qw * qz); Rb[4] = 1.f - 2.f * (qx * qx + qz * qz); Rb[5] = 2.f * (qy * qz - qw * qx);
    Rb[6] = 2.f * (qx * qz - qw * qy); Rb[7] = 2.f * (qy * qz + qw * qx); Rb[8] = 1.f - 2.f * (qx * qx + qy * qy);
    float tbx = af[4], tby = af[5], tbz = af[6];

    const int aa = aatype[r];
    const float* mf = dframes + (long)aa * 128;  // [8][4][4]

    // rot_f[g] = frame_rot @ x-rot(angle g); trans_f[g] = frame_trans
    float RA[8][9], TA[8][3];
#pragma unroll
    for (int g = 0; g < 8; ++g) {
      const float* F = mf + g * 16;
      float cg = cs[g], sg = sn[g];
#pragma unroll
      for (int rr = 0; rr < 3; ++rr) {
        float R0 = F[rr * 4 + 0], R1 = F[rr * 4 + 1], R2 = F[rr * 4 + 2];
        RA[g][rr * 3 + 0] = R0;
        RA[g][rr * 3 + 1] = R1 * cg + R2 * sg;
        RA[g][rr * 3 + 2] = -R1 * sg + R2 * cg;
        TA[g][rr] = F[rr * 4 + 3];
      }
    }
    // compose chain for groups 5,6,7 (chi2..chi4 stack on previous)
#pragma unroll
    for (int g = 5; g < 8; ++g) {
      float Rn[9], Tn[3];
      mat3mul(Rn, RA[g - 1], RA[g]);
#pragma unroll
      for (int rr = 0; rr < 3; ++rr)
        Tn[rr] = RA[g - 1][rr * 3 + 0] * TA[g][0] + RA[g - 1][rr * 3 + 1] * TA[g][1] +
                 RA[g - 1][rr * 3 + 2] * TA[g][2] + TA[g - 1][rr];
#pragma unroll
      for (int k = 0; k < 9; ++k) RA[g][k] = Rn[k];
#pragma unroll
      for (int k = 0; k < 3; ++k) TA[g][k] = Tn[k];
    }
    // into global frame; write rot_g / trans_g
#pragma unroll
    for (int g = 0; g < 8; ++g) {
      float RG[9], TG[3];
      mat3mul(RG, Rb, RA[g]);
      TG[0] = Rb[0] * TA[g][0] + Rb[1] * TA[g][1] + Rb[2] * TA[g][2] + tbx;
      TG[1] = Rb[3] * TA[g][0] + Rb[4] * TA[g][1] + Rb[5] * TA[g][2] + tby;
      TG[2] = Rb[6] * TA[g][0] + Rb[7] * TA[g][1] + Rb[8] * TA[g][2] + tbz;
      float* rg = out + o_rot + (long)r * 72 + g * 9;
#pragma unroll
      for (int k = 0; k < 9; ++k) rg[k] = RG[k];
      float* tg = out + o_trans + (long)r * 24 + g * 3;
      tg[0] = TG[0]; tg[1] = TG[1]; tg[2] = TG[2];
    }
    // atom positions: dynamic group gather re-reads what we just wrote
    // (same-wave store->load to same address is kept in order per ISA).
#pragma unroll
    for (int a = 0; a < 14; ++a) {
      int idx = aa * 14 + a;
      int g = gidx[idx];
      const float* RG = out + o_rot + (long)r * 72 + g * 9;
      const float* TG = out + o_trans + (long)r * 24 + g * 3;
      const float* L = lpos + (long)idx * 3;
      float msk = amask[idx];
      float px = (RG[0] * L[0] + RG[1] * L[1] + RG[2] * L[2] + TG[0]) * msk;
      float py = (RG[3] * L[0] + RG[4] * L[1] + RG[5] * L[2] + TG[1]) * msk;
      float pz = (RG[6] * L[0] + RG[7] * L[1] + RG[8] * L[2] + TG[2]) * msk;
      float* p = out + o_pos + (long)r * 42 + a * 3;
      p[0] = px; p[1] = py; p[2] = pz;
    }
  }
}

extern "C" void kernel_launch(void* const* d_in, const int* in_sizes, int n_in,
                              void* d_out, int out_size, void* d_ws, size_t ws_size,
                              hipStream_t stream) {
  const float* affine = (const float*)d_in[0];
  const float* rep0   = (const float*)d_in[1];
  const float* rep1   = (const float*)d_in[2];
  const int*   aatype = (const int*)d_in[3];
  const float* w_in0  = (const float*)d_in[4];
  const float* b_in0  = (const float*)d_in[5];
  const float* w_in1  = (const float*)d_in[6];
  const float* b_in1  = (const float*)d_in[7];
  const float* Wr1    = (const float*)d_in[8];
  const float* br1    = (const float*)d_in[9];
  const float* Wr2    = (const float*)d_in[10];
  const float* br2    = (const float*)d_in[11];
  const float* w_ang  = (const float*)d_in[12];
  const float* b_ang  = (const float*)d_in[13];
  const float* dframes = (const float*)d_in[14];
  const int*   gidx   = (const int*)d_in[15];
  const float* amask  = (const float*)d_in[16];
  const float* lpos   = (const float*)d_in[17];
  float* out = (float*)d_out;
  _Float16* ws = (_Float16*)d_ws;

  const int n_res = in_sizes[1] / N_MSA;

  prep_weights_kernel<<<(WS_TOTAL + 255) / 256, 256, 0, stream>>>(
      w_in0, w_in1, Wr1, Wr2, w_ang, ws);
  sidechain_fused_kernel<<<n_res / 128, 256, 0, stream>>>(
      affine, rep0, rep1, aatype, b_in0, b_in1, br1, br2, b_ang,
      dframes, gidx, amask, lpos, ws, out, n_res);
}
